// Net_22101901705332
// MI455X (gfx1250) — matrix operand order repaired and checked by reading, compiled, not run on hardware
//
#include <hip/hip_runtime.h>
#include <cstdint>
#include <cstddef>

// ---------------------------------------------------------------------------
// GCN forward for MI455X (gfx1250, wave32, WMMA + TDM).
//   h = relu(Agg(x @ W1) + b1);  o = Agg(h @ W2) + b2;  out = log_softmax(o)
// GEMMs run on v_wmma_f32_16x16x32_bf16 (bf16 in, fp32 accum).
// GEMM2 stages its A-tile with tensor_load_to_lds (TDM), using TDM padding to
// produce the bank-conflict-free LDS layout directly.
// Sparse aggregation: wave-per-edge gather (bf16 rows) + fp32 L2 atomics.
// ---------------------------------------------------------------------------

#define N_NODES 100000
#define N_EDGES 1600000
#define NFEAT   512
#define NHID    256
#define NCLASS  64

typedef __bf16 bf16_t;
typedef __attribute__((ext_vector_type(16))) __bf16    v16bf;
typedef __attribute__((ext_vector_type(8)))  __bf16    v8bf;
typedef __attribute__((ext_vector_type(8)))  float     v8f;
typedef __attribute__((ext_vector_type(4)))  uint32_t  u32x4;
typedef __attribute__((ext_vector_type(8)))  uint32_t  u32x8;

__device__ __forceinline__ bf16_t f2bf(float x) { return (bf16_t)x; }

__device__ __forceinline__ v8f wmma_bf16(v16bf a, v16bf b, v8f c) {
  return __builtin_amdgcn_wmma_f32_16x16x32_bf16(false, a, false, b,
                                                 (short)0, c, false, false);
}

// ---------------------------------------------------------------- utilities
__global__ void zero_f32(float* __restrict__ p, long long n) {
  long long i = (long long)blockIdx.x * blockDim.x + threadIdx.x;
  long long stride = (long long)gridDim.x * blockDim.x;
  for (; i < n; i += stride) p[i] = 0.0f;
}

// deg[dst] += ew  (self-loop weight 1.0 added analytically in dinv_kernel)
__global__ void degree_kernel(const int* __restrict__ dst,
                              const float* __restrict__ ew,
                              float* __restrict__ deg) {
  int e = blockIdx.x * blockDim.x + threadIdx.x;
  if (e < N_EDGES) atomicAdd(deg + dst[e], ew[e]);
}

// in-place: deg -> D^{-1/2}
__global__ void dinv_kernel(float* __restrict__ deg) {
  int i = blockIdx.x * blockDim.x + threadIdx.x;
  if (i < N_NODES) {
    float d = deg[i] + 1.0f;           // + self-loop weight
    deg[i] = (d > 0.0f) ? rsqrtf(d) : 0.0f;
  }
}

__global__ void norm_kernel(const int* __restrict__ src, const int* __restrict__ dst,
                            const float* __restrict__ ew, const float* __restrict__ dinv,
                            float* __restrict__ normv) {
  int e = blockIdx.x * blockDim.x + threadIdx.x;
  if (e < N_EDGES) normv[e] = dinv[src[e]] * ew[e] * dinv[dst[e]];
}

// WT[c*K + k] = bf16(W[k*C + c])  — column-major bf16 weights so each lane's
// WMMA B-fragment is one contiguous 32B v16bf load.
__global__ void transpose_bf16(const float* __restrict__ W, bf16_t* __restrict__ WT,
                               int K, int C) {
  int idx = blockIdx.x * blockDim.x + threadIdx.x;
  if (idx >= K * C) return;
  int c = idx / K, k = idx - c * K;
  WT[idx] = f2bf(W[(size_t)k * C + c]);
}

// ------------------------------------------------------------------- GEMM 1
// s1[100000,256] = bf16(x[100000,512]) @ W1 ; block = 32 rows x 256 cols.
// Each wave: 2x2 grid of 16x16 tiles -> every B fragment feeds 2 WMMAs,
// halving W1T L2 traffic vs 16-row blocks. Unroll capped at 4 to stay under
// 256 VGPRs (avoids s_set_vgpr_msb juggling, keeps occupancy up).
__global__ __launch_bounds__(256) void gemm1_wmma(const float* __restrict__ x,
                                                  const bf16_t* __restrict__ W1T,
                                                  bf16_t* __restrict__ s1) {
  __shared__ bf16_t As[32][NFEAT + 8];   // +8 bf16 pad -> conflict-free frag reads
  const int row0 = blockIdx.x * 32;

  for (int idx = threadIdx.x; idx < 32 * NFEAT; idx += 256) {
    int r = idx >> 9, c = idx & (NFEAT - 1);
    As[r][c] = f2bf(x[(size_t)(row0 + r) * NFEAT + c]);
  }
  __syncthreads();

  const int wid  = threadIdx.x >> 5;
  const int lane = threadIdx.x & 31;
  const int c0   = wid * 32;                 // 8 waves x 32 cols = 256
  const int ar0  = lane & 15;                // A row, tile 0
  const int ar1  = ar0 + 16;                 // A row, tile 1
  const int kh   = (lane >> 4) * 8;          // A-frag K-half select
  const int bcol = c0 + (lane & 15);
  const int kb   = (lane >> 4) * 16;         // B-frag K-half select

  v8f acc00 = {}, acc01 = {}, acc10 = {}, acc11 = {};
#pragma unroll 4
  for (int kk = 0; kk < NFEAT / 32; ++kk) {
    const int k0 = kk * 32;
    union { uint32_t u[8]; v16bf v; } a0, a1;
#pragma unroll
    for (int vv = 0; vv < 8; ++vv) {
      int k = k0 + 2 * kh + 2 * vv;
      a0.u[vv] = *(const uint32_t*)&As[ar0][k];
      a1.u[vv] = *(const uint32_t*)&As[ar1][k];
    }
    v16bf b0 = *(const v16bf*)(W1T + (size_t)bcol * NFEAT + k0 + kb);
    v16bf b1 = *(const v16bf*)(W1T + (size_t)(bcol + 16) * NFEAT + k0 + kb);
    acc00 = wmma_bf16(a0.v, b0, acc00);
    acc01 = wmma_bf16(a0.v, b1, acc01);
    acc10 = wmma_bf16(a1.v, b0, acc10);
    acc11 = wmma_bf16(a1.v, b1, acc11);
  }
  const int m0 = (lane >> 4) * 8, n = lane & 15;
#pragma unroll
  for (int vv = 0; vv < 8; ++vv) {
    size_t r0w = (size_t)(row0 + m0 + vv) * NHID;
    size_t r1w = (size_t)(row0 + 16 + m0 + vv) * NHID;
    s1[r0w + c0 + n]      = f2bf(acc00[vv]);
    s1[r0w + c0 + 16 + n] = f2bf(acc01[vv]);
    s1[r1w + c0 + n]      = f2bf(acc10[vv]);
    s1[r1w + c0 + 16 + n] = f2bf(acc11[vv]);
  }
}

// ------------------------------------------------------------------- GEMM 2
// s2[100000,64] = h[100000,256](bf16) @ W2 ; block = 32 rows x 64 cols.
// A-tile staged by the Tensor Data Mover: one tensor_load_to_lds pulls the
// 32x256 bf16 tile; TDM pad (4 DWORDs every 128 DWORDs) writes the padded
// 264-element LDS rows that make fragment reads bank-conflict-free.
__global__ __launch_bounds__(256) void gemm2_wmma(const bf16_t* __restrict__ h,
                                                  const bf16_t* __restrict__ W2T,
                                                  bf16_t* __restrict__ s2) {
  __shared__ bf16_t As[32][NHID + 8];
  const int row0 = blockIdx.x * 32;

  if (__builtin_amdgcn_readfirstlane(threadIdx.x >> 5) == 0) {  // wave 0 only
    uint64_t gaddr = (uint64_t)(uintptr_t)(const void*)(h + (size_t)row0 * NHID);
    uint32_t laddr = (uint32_t)(uintptr_t)(void*)&As[0][0];
    // D# group 0: count=1 | lds_addr | global_addr[56:0] | type=2
    u32x4 g0 = { 0x1u,
                 laddr,
                 (uint32_t)gaddr,
                 (uint32_t)((gaddr >> 32) & 0x01ffffffu) | (2u << 30) };
    // D# group 1: data_size=2B, pad_enable, pad_interval=6 (128 DW),
    // pad_amount=3 (4 DW); tensor 256 x 100000, tile 256 x 32, stride 256.
    u32x8 g1 = { (1u << 16) | (1u << 20) | (6u << 22) | (3u << 25),
                 ((uint32_t)NHID & 0xffffu) << 16,
                 ((uint32_t)N_NODES & 0xffffu) << 16,
                 ((uint32_t)N_NODES >> 16) | (((uint32_t)NHID & 0xffffu) << 16),
                 32u,                       // tile_dim1 = 32 rows
                 (uint32_t)NHID,            // tensor_dim0_stride = 256
                 0u, 0u };
    asm volatile("tensor_load_to_lds %0, %1" :: "s"(g0), "s"(g1) : "memory");
  }
  __builtin_amdgcn_s_wait_tensorcnt(0);
  __syncthreads();

  const int wid  = threadIdx.x >> 5;
  const int lane = threadIdx.x & 31;
  const int rb   = (wid >> 2) * 16;          // 2 row tiles
  const int cb   = (wid & 3) * 16;           // 4 col tiles
  const int arow = rb + (lane & 15);
  const int kh   = (lane >> 4) * 8;
  const int bcol = cb + (lane & 15);
  const int kb   = (lane >> 4) * 16;

  v8f acc = {};
#pragma unroll 4
  for (int kk = 0; kk < NHID / 32; ++kk) {
    const int k0 = kk * 32;
    union { uint32_t u[8]; v16bf v; } a;
#pragma unroll
    for (int vv = 0; vv < 8; ++vv) {
      int k = k0 + 2 * kh + 2 * vv;
      a.u[vv] = *(const uint32_t*)&As[arow][k];
    }
    v16bf b = *(const v16bf*)(W2T + (size_t)bcol * NHID + k0 + kb);
    acc = wmma_bf16(a.v, b, acc);
  }
  const int m0 = (lane >> 4) * 8, n = lane & 15;
#pragma unroll
  for (int vv = 0; vv < 8; ++vv)
    s2[(size_t)(row0 + rb + m0 + vv) * NCLASS + cb + n] = f2bf(acc[vv]);
}

// ------------------------------------------------------------- aggregation
// One wave32 per edge: lane l covers 8 features -> coalesced 1KB-row gather
// (bf16 = 512B traffic) + fp32 atomics into L2-resident accumulator.
__global__ __launch_bounds__(256) void agg1_edges(const int* __restrict__ src,
                                                  const int* __restrict__ dst,
                                                  const float* __restrict__ normv,
                                                  const bf16_t* __restrict__ s1,
                                                  float* __restrict__ agg) {
  int gw   = (blockIdx.x * 256 + threadIdx.x) >> 5;
  int lane = threadIdx.x & 31;
  if (gw >= N_EDGES) return;
  int s = src[gw], d = dst[gw];
  float w = normv[gw];
  v8bf vals = *(const v8bf*)(s1 + (size_t)s * NHID + lane * 8);
  float* arow = agg + (size_t)d * NHID + lane * 8;
#pragma unroll
  for (int j = 0; j < 8; ++j) atomicAdd(arow + j, w * (float)vals[j]);
}

__global__ __launch_bounds__(256) void agg2_edges(const int* __restrict__ src,
                                                  const int* __restrict__ dst,
                                                  const float* __restrict__ normv,
                                                  const bf16_t* __restrict__ s2,
                                                  float* __restrict__ agg) {
  int gw   = (blockIdx.x * 256 + threadIdx.x) >> 5;
  int lane = threadIdx.x & 31;
  if (gw >= N_EDGES) return;
  int s = src[gw], d = dst[gw];
  float w = normv[gw];
  const bf16_t* srow = s2 + (size_t)s * NCLASS;
  float* arow = agg + (size_t)d * NCLASS;
  atomicAdd(arow + lane,      w * (float)srow[lane]);
  atomicAdd(arow + lane + 32, w * (float)srow[lane + 32]);
}

// h = relu(agg1 + dinv^2 * s1 + b1)  (self-loop folded in), stored as bf16
__global__ void finish1(const float* __restrict__ agg, const bf16_t* __restrict__ s1,
                        const float* __restrict__ dinv, const float* __restrict__ b1,
                        bf16_t* __restrict__ h) {
  long long idx = (long long)blockIdx.x * blockDim.x + threadIdx.x;
  if (idx >= (long long)N_NODES * NHID) return;
  int row = (int)(idx >> 8), f = (int)(idx & (NHID - 1));
  float dv = dinv[row];
  float v = agg[idx] + dv * dv * (float)s1[idx] + b1[f];
  h[idx] = f2bf(fmaxf(v, 0.0f));
}

// o = agg2 + dinv^2 * s2 + b2, then in-wave log_softmax over 64 classes
__global__ __launch_bounds__(256) void finish2_softmax(const float* __restrict__ agg,
                                                       const bf16_t* __restrict__ s2,
                                                       const float* __restrict__ dinv,
                                                       const float* __restrict__ b2,
                                                       float* __restrict__ out) {
  int row  = (blockIdx.x * 256 + threadIdx.x) >> 5;
  int lane = threadIdx.x & 31;
  if (row >= N_NODES) return;
  float dv = dinv[row], dv2 = dv * dv;
  size_t base = (size_t)row * NCLASS;
  float o0 = agg[base + lane]      + dv2 * (float)s2[base + lane]      + b2[lane];
  float o1 = agg[base + lane + 32] + dv2 * (float)s2[base + lane + 32] + b2[lane + 32];
  float m = fmaxf(o0, o1);
#pragma unroll
  for (int off = 16; off > 0; off >>= 1) m = fmaxf(m, __shfl_xor(m, off));
  float sum = __expf(o0 - m) + __expf(o1 - m);
#pragma unroll
  for (int off = 16; off > 0; off >>= 1) sum += __shfl_xor(sum, off);
  float lse = m + __logf(sum);
  out[base + lane]      = o0 - lse;
  out[base + lane + 32] = o1 - lse;
}

// ------------------------------------------------------------------ launch
extern "C" void kernel_launch(void* const* d_in, const int* in_sizes, int n_in,
                              void* d_out, int out_size, void* d_ws, size_t ws_size,
                              hipStream_t stream) {
  (void)in_sizes; (void)n_in; (void)out_size; (void)ws_size;
  const float* x  = (const float*)d_in[0];
  const int*   ei = (const int*)  d_in[1];
  const float* ew = (const float*)d_in[2];
  const float* W1 = (const float*)d_in[3];
  const float* b1 = (const float*)d_in[4];
  const float* W2 = (const float*)d_in[5];
  const float* b2 = (const float*)d_in[6];
  const int* src = ei;            // edge_index[0]
  const int* dst = ei + N_EDGES;  // edge_index[1]

  // workspace carve-out (~212 MB peak; layer-2 scratch aliases dead agg1)
  char* p = (char*)d_ws;
  auto alloc = [&](size_t bytes) {
    char* q = p; p += (bytes + 255) & ~(size_t)255; return q;
  };
  float*  deg  = (float*) alloc(sizeof(float)  * N_NODES);                  // -> dinv
  float*  nrm  = (float*) alloc(sizeof(float)  * N_EDGES);
  bf16_t* W1T  = (bf16_t*)alloc(sizeof(bf16_t) * NFEAT * NHID);
  bf16_t* W2T  = (bf16_t*)alloc(sizeof(bf16_t) * NHID * NCLASS);
  bf16_t* s1   = (bf16_t*)alloc(sizeof(bf16_t) * (size_t)N_NODES * NHID);
  float*  agg1 = (float*) alloc(sizeof(float)  * (size_t)N_NODES * NHID);   // 102.4 MB
  bf16_t* hbuf = (bf16_t*)alloc(sizeof(bf16_t) * (size_t)N_NODES * NHID);
  bf16_t* s2   = (bf16_t*)agg1;                          // 12.8 MB, reuse
  float*  agg2 = (float*)((char*)agg1 + (40u << 20));    // 25.6 MB, disjoint reuse

  dim3 B(256);
  zero_f32      <<<(N_NODES + 255) / 256, B, 0, stream>>>(deg, N_NODES);
  degree_kernel <<<(N_EDGES + 255) / 256, B, 0, stream>>>(dst, ew, deg);
  dinv_kernel   <<<(N_NODES + 255) / 256, B, 0, stream>>>(deg);
  norm_kernel   <<<(N_EDGES + 255) / 256, B, 0, stream>>>(src, dst, ew, deg, nrm);
  transpose_bf16<<<(NFEAT * NHID  + 255) / 256, B, 0, stream>>>(W1, W1T, NFEAT, NHID);
  transpose_bf16<<<(NHID * NCLASS + 255) / 256, B, 0, stream>>>(W2, W2T, NHID, NCLASS);

  gemm1_wmma<<<N_NODES / 32, B, 0, stream>>>(x, W1T, s1);                 // 3125 blocks
  zero_f32  <<<100000, B, 0, stream>>>(agg1, (long long)N_NODES * NHID);
  agg1_edges<<<(N_EDGES + 7) / 8, B, 0, stream>>>(src, dst, nrm, s1, agg1);
  finish1   <<<(int)(((long long)N_NODES * NHID + 255) / 256), B, 0, stream>>>(
      agg1, s1, deg, b1, hbuf);

  gemm2_wmma<<<N_NODES / 32, B, 0, stream>>>(hbuf, W2T, s2);              // 3125 blocks
  zero_f32  <<<100000, B, 0, stream>>>(agg2, (long long)N_NODES * NCLASS);
  agg2_edges<<<(N_EDGES + 7) / 8, B, 0, stream>>>(src, dst, nrm, s2, agg2);
  finish2_softmax<<<(N_NODES + 7) / 8, B, 0, stream>>>(agg2, s2, deg, b2,
                                                       (float*)d_out);
}
